// MoE_40312563040936
// MI455X (gfx1250) — matrix operand order repaired and checked by reading, not compiled, not run on hardware
//
#include <hip/hip_runtime.h>
#include <hip/hip_bf16.h>
#include <math.h>

// ---------------- problem constants ----------------
constexpr int S = 8192;     // tokens (4*2048)
constexpr int H = 2048;     // hidden dim (M)
constexpr int F = 8192;     // ffn dim
constexpr int E = 8;        // experts
constexpr int C = 1024;     // capacity per expert (S/E * 1.0)

// ---------------- vector types ----------------
typedef __attribute__((ext_vector_type(16))) __bf16 v16bf;
typedef __attribute__((ext_vector_type(8)))  __bf16 v8bf;
typedef __attribute__((ext_vector_type(2)))  __bf16 v2bf;
typedef __attribute__((ext_vector_type(8)))  float  v8f;
typedef __attribute__((ext_vector_type(4)))  unsigned int v4u;
typedef __attribute__((ext_vector_type(8)))  int v8i_t;
typedef __attribute__((ext_vector_type(4)))  int v4i_t;

#if defined(__has_builtin)
#if __has_builtin(__builtin_amdgcn_tensor_load_to_lds) && __has_builtin(__builtin_amdgcn_s_wait_tensorcnt)
#define USE_TDM 1
#endif
#endif

// pack two f32 -> packed bf16 (RNE); lowers to v_cvt_pk_bf16_f32 when available
static __device__ __forceinline__ unsigned int pk_bf16(float lo, float hi) {
#if defined(__has_builtin) && __has_builtin(__builtin_amdgcn_cvt_pk_bf16_f32)
    v2bf r = __builtin_amdgcn_cvt_pk_bf16_f32(lo, hi);
    return __builtin_bit_cast(unsigned int, r);
#else
    v2bf r; r[0] = (__bf16)lo; r[1] = (__bf16)hi;
    return __builtin_bit_cast(unsigned int, r);
#endif
}

static __device__ __forceinline__ unsigned short f2bf(float f) {
    v2bf r; r[0] = (__bf16)f; r[1] = (__bf16)0.0f;
    return (unsigned short)(__builtin_bit_cast(unsigned int, r) & 0xFFFFu);
}

static __device__ __forceinline__ v16bf mk16(v8bf lo, v8bf hi) {
    v16bf r;
#pragma unroll
    for (int i = 0; i < 8; ++i) { r[i] = lo[i]; r[i + 8] = hi[i]; }
    return r;
}

// tanh-approx gelu via sigmoid: x * sigmoid(1.5957691*(x + 0.044715 x^3))
static __device__ __forceinline__ float gelu_tanh(float x) {
    float z = 1.5957691216057308f * (x + 0.044715f * x * x * x);
    float s = __frcp_rn(1.0f + __expf(-z));
    return x * s;
}

// ---------------- kernel 0: zero out + stats ----------------
__global__ void moe_zero_kernel(float* __restrict__ out, int n,
                                unsigned int* __restrict__ stats /*16 dwords*/) {
    int i = blockIdx.x * blockDim.x + threadIdx.x;
    if (i < n) out[i] = 0.0f;
    if (i < 16) stats[i] = 0u;
}

// ---------------- kernel 1: top-1 gating ----------------
__global__ void moe_gate_kernel(const float* __restrict__ x,
                                const float* __restrict__ wg,
                                int* __restrict__ idx1,
                                float* __restrict__ gate_max,
                                float* __restrict__ stats) {
    int lane = threadIdx.x & 31;
    int wid  = threadIdx.x >> 5;
    int s    = blockIdx.x * 8 + wid;
    if (s >= S) return;

    float acc[E];
#pragma unroll
    for (int e = 0; e < E; ++e) acc[e] = 0.0f;

    const float* xrow = x + (size_t)s * H;
#pragma unroll 4
    for (int i = 0; i < H / 32; ++i) {
        int m = i * 32 + lane;
        float xv = xrow[m];
        const float4* wr = (const float4*)(wg + (size_t)m * E);
        float4 w0 = wr[0], w1 = wr[1];
        acc[0] += xv * w0.x; acc[1] += xv * w0.y;
        acc[2] += xv * w0.z; acc[3] += xv * w0.w;
        acc[4] += xv * w1.x; acc[5] += xv * w1.y;
        acc[6] += xv * w1.z; acc[7] += xv * w1.w;
    }
#pragma unroll
    for (int off = 16; off > 0; off >>= 1) {
#pragma unroll
        for (int e = 0; e < E; ++e) acc[e] += __shfl_xor(acc[e], off, 32);
    }

    float mx = acc[0]; int idx = 0;
#pragma unroll
    for (int e = 1; e < E; ++e) { if (acc[e] > mx) { mx = acc[e]; idx = e; } }
    float sum = 0.0f;
    float g[E];
#pragma unroll
    for (int e = 0; e < E; ++e) { g[e] = __expf(acc[e] - mx); sum += g[e]; }
    float inv = 1.0f / sum;

    if (lane == 0) {
        idx1[s] = idx;
        gate_max[s] = inv;
        atomicAdd((int*)(stats + 8) + idx, 1);
    }
    if (lane < E) atomicAdd(&stats[lane], g[lane] * inv);
}

// ---------------- kernel 2: capacity scan ----------------
__global__ __launch_bounds__(1024)
void moe_scan_kernel(const int* __restrict__ idx1,
                     const float* __restrict__ gate_max,
                     int* __restrict__ tok_of_slot,
                     float* __restrict__ gate_of_slot) {
    __shared__ int sh_wtot[32];
    __shared__ int sh_woff[32];
    __shared__ int sh_base;
    __shared__ int sh_ctot;

    int e    = blockIdx.x;
    int tid  = threadIdx.x;
    int lane = tid & 31;
    int wid  = tid >> 5;

    if (tid < C) { tok_of_slot[e * C + tid] = -1; gate_of_slot[e * C + tid] = 0.0f; }
    if (tid == 0) sh_base = 0;
    __syncthreads();

    for (int chunk = 0; chunk < S / 1024; ++chunk) {
        int s = chunk * 1024 + tid;
        int flag = (idx1[s] == e) ? 1 : 0;
        unsigned int b32 = (unsigned int)__ballot(flag);
        int wprefix = __popc(b32 & ((1u << lane) - 1u));
        if (lane == 0) sh_wtot[wid] = __popc(b32);
        __syncthreads();
        if (tid < 32) {
            int v = sh_wtot[tid], orig = v;
#pragma unroll
            for (int off = 1; off < 32; off <<= 1) {
                int n = __shfl_up(v, off, 32);
                if (lane >= off) v += n;
            }
            sh_woff[tid] = v - orig;
            if (tid == 31) sh_ctot = v;
        }
        __syncthreads();
        int slot = sh_base + sh_woff[wid] + wprefix;
        if (flag && slot < C) {
            tok_of_slot[e * C + slot]  = s;
            gate_of_slot[e * C + slot] = gate_max[s];
        }
        __syncthreads();
        if (tid == 0) sh_base += sh_ctot;
        __syncthreads();
    }
}

// ---------------- GEMM tile geometry ----------------
constexpr int BM = 128, BN = 128, BK = 32;
constexpr int LDT = 40;   // ushort pitch (80B rows: 64B data + 16B pad)

// fragment loads shared by both GEMMs
static __device__ __forceinline__ void load_frags(const unsigned short* As,
                                                  const unsigned short* Bs,
                                                  int wm, int wn, int grp, int lr,
                                                  v16bf afrag[2], v16bf bfrag[4]) {
#pragma unroll
    for (int i = 0; i < 2; ++i) {
        const unsigned short* base = &As[(wm * 32 + i * 16 + lr) * LDT];
        afrag[i] = mk16(*(const v8bf*)(base + 8 * grp), *(const v8bf*)(base + 16 + 8 * grp));
    }
#pragma unroll
    for (int j = 0; j < 4; ++j) {
        const unsigned short* base = &Bs[(wn * 64 + j * 16 + lr) * LDT];
        bfrag[j] = mk16(*(const v8bf*)(base + 8 * grp), *(const v8bf*)(base + 16 + 8 * grp));
    }
}

// ---------------- kernel 3: GEMM1  h = gelu(gather(x) @ w1 + b1) ----------------
__global__ __launch_bounds__(256)
void moe_gemm1_kernel(const float* __restrict__ x,
                      const float* __restrict__ w1,
                      const float* __restrict__ b1,
                      const int* __restrict__ tok_of_slot,
                      unsigned short* __restrict__ hbuf) {
    __shared__ __attribute__((aligned(64))) unsigned short As[BM * LDT];
    __shared__ __attribute__((aligned(64))) unsigned short Bs[BN * LDT];

    const int e  = blockIdx.z;
    const int m0 = blockIdx.y * BM;
    const int n0 = blockIdx.x * BN;
    const int tid  = threadIdx.x;
    const int lane = tid & 31;
    const int wid  = tid >> 5;
    const int wm = wid & 3, wn = wid >> 2;
    const int grp = lane >> 4, lr = lane & 15;

    v8f acc[2][4];
#pragma unroll
    for (int i = 0; i < 2; ++i)
#pragma unroll
        for (int j = 0; j < 4; ++j)
#pragma unroll
            for (int k = 0; k < 8; ++k) acc[i][j][k] = 0.0f;

    // A staging: thread -> (row, 16-wide half);  gathered token row
    const int arow  = tid >> 1;
    const int ahalf = tid & 1;
    const int token = tok_of_slot[e * C + m0 + arow];
    const float* asrc0 = (token >= 0) ? (x + (size_t)token * H + ahalf * 16) : nullptr;

    // B staging: thread -> (column, 16-deep K half); coalesced dword loads
    const int ncol  = tid & 127;
    const int khalf = tid >> 7;
    const float* bsrc0 = w1 + ((size_t)e * H + khalf * 16) * F + n0 + ncol;

    float4 apre[4];
    float  bpre[16];

    auto load_a = [&](int kt) {
        if (asrc0) {
            const float4* p = (const float4*)(asrc0 + kt * BK);
#pragma unroll
            for (int q = 0; q < 4; ++q) apre[q] = p[q];
        } else {
#pragma unroll
            for (int q = 0; q < 4; ++q) apre[q] = make_float4(0.f, 0.f, 0.f, 0.f);
        }
    };
    auto load_b = [&](int kt) {
        const float* p = bsrc0 + (size_t)kt * BK * F;
#pragma unroll
        for (int r = 0; r < 16; ++r) bpre[r] = p[(size_t)r * F];
    };
    auto store_ab = [&]() {
        uint4 u0, u1;
        u0.x = pk_bf16(apre[0].x, apre[0].y); u0.y = pk_bf16(apre[0].z, apre[0].w);
        u0.z = pk_bf16(apre[1].x, apre[1].y); u0.w = pk_bf16(apre[1].z, apre[1].w);
        u1.x = pk_bf16(apre[2].x, apre[2].y); u1.y = pk_bf16(apre[2].z, apre[2].w);
        u1.z = pk_bf16(apre[3].x, apre[3].y); u1.w = pk_bf16(apre[3].z, apre[3].w);
        uint4* ad = (uint4*)(&As[arow * LDT + ahalf * 16]);
        ad[0] = u0; ad[1] = u1;
        uint4 v0, v1;
        v0.x = pk_bf16(bpre[0], bpre[1]);   v0.y = pk_bf16(bpre[2], bpre[3]);
        v0.z = pk_bf16(bpre[4], bpre[5]);   v0.w = pk_bf16(bpre[6], bpre[7]);
        v1.x = pk_bf16(bpre[8], bpre[9]);   v1.y = pk_bf16(bpre[10], bpre[11]);
        v1.z = pk_bf16(bpre[12], bpre[13]); v1.w = pk_bf16(bpre[14], bpre[15]);
        uint4* bd = (uint4*)(&Bs[ncol * LDT + khalf * 16]);
        bd[0] = v0; bd[1] = v1;
    };

    load_a(0); load_b(0);
    for (int kt = 0; kt < H / BK; ++kt) {
        __syncthreads();
        store_ab();
        __syncthreads();
        if (kt + 1 < H / BK) { load_a(kt + 1); load_b(kt + 1); }   // overlap with WMMA

        v16bf afrag[2], bfrag[4];
        load_frags(As, Bs, wm, wn, grp, lr, afrag, bfrag);
#pragma unroll
        for (int i = 0; i < 2; ++i)
#pragma unroll
            for (int j = 0; j < 4; ++j)
                acc[i][j] = __builtin_amdgcn_wmma_f32_16x16x32_bf16(
                    false, afrag[i], false, bfrag[j], (short)0, acc[i][j], false, false);
    }

    // epilogue: bias + gelu -> bf16 h
#pragma unroll
    for (int i = 0; i < 2; ++i)
#pragma unroll
        for (int j = 0; j < 4; ++j) {
            int col = n0 + wn * 64 + j * 16 + lr;
            float bias = b1[e * F + col];
#pragma unroll
            for (int rj = 0; rj < 8; ++rj) {
                int row = m0 + wm * 32 + i * 16 + grp * 8 + rj;
                float v = gelu_tanh(acc[i][j][rj] + bias);
                hbuf[((size_t)(e * C + row)) * F + col] = f2bf(v);
            }
        }
}

// ---------------- kernel 4: GEMM2 + combine scatter ----------------
__global__ __launch_bounds__(256)
void moe_gemm2_kernel(const unsigned short* __restrict__ hbuf,
                      const float* __restrict__ w2,
                      const float* __restrict__ b2,
                      const int* __restrict__ tok_of_slot,
                      const float* __restrict__ gate_of_slot,
                      float* __restrict__ out) {
    __shared__ __attribute__((aligned(64))) unsigned short As[BM * LDT];
    __shared__ __attribute__((aligned(64))) unsigned short Bs[BN * LDT];

    const int e  = blockIdx.z;
    const int m0 = blockIdx.y * BM;
    const int n0 = blockIdx.x * BN;
    const int tid  = threadIdx.x;
    const int lane = tid & 31;
    const int wid  = tid >> 5;
    const int wm = wid & 3, wn = wid >> 2;
    const int grp = lane >> 4, lr = lane & 15;

    v8f acc[2][4];
#pragma unroll
    for (int i = 0; i < 2; ++i)
#pragma unroll
        for (int j = 0; j < 4; ++j)
#pragma unroll
            for (int k = 0; k < 8; ++k) acc[i][j][k] = 0.0f;

    // B staging: coalesced dword loads by (column, K-half)
    const int ncol  = tid & 127;
    const int khalf = tid >> 7;
    const float* bsrc0 = w2 + ((size_t)e * F + khalf * 16) * H + n0 + ncol;
    float bpre[16];

    auto load_b = [&](int kt) {
        const float* p = bsrc0 + (size_t)kt * BK * H;
#pragma unroll
        for (int r = 0; r < 16; ++r) bpre[r] = p[(size_t)r * H];
    };
    auto store_b = [&]() {
        uint4 v0, v1;
        v0.x = pk_bf16(bpre[0], bpre[1]);   v0.y = pk_bf16(bpre[2], bpre[3]);
        v0.z = pk_bf16(bpre[4], bpre[5]);   v0.w = pk_bf16(bpre[6], bpre[7]);
        v1.x = pk_bf16(bpre[8], bpre[9]);   v1.y = pk_bf16(bpre[10], bpre[11]);
        v1.z = pk_bf16(bpre[12], bpre[13]); v1.w = pk_bf16(bpre[14], bpre[15]);
        uint4* bd = (uint4*)(&Bs[ncol * LDT + khalf * 16]);
        bd[0] = v0; bd[1] = v1;
    };

#if USE_TDM
    // TDM descriptor for the bf16 A tile: 128 rows x 32 elems (64B),
    // LDS pitch 80B via pad (16 DWORDs data, 4 DWORDs pad).
    const unsigned int lds_addr = (unsigned int)(uintptr_t)(&As[0]);
    const unsigned short* atile0 = hbuf + ((size_t)(e * C + m0)) * F;
    v8i_t g1;
    g1[0] = (int)((1u << 16) | (1u << 20) | (3u << 22) | (3u << 25)); // data_size=2B, pad_en, int=16dw, amt=4dw
    g1[1] = (int)((unsigned)(F & 0xFFFF) << 16);   // tensor_dim0 = F
    g1[2] = (int)(128u << 16);                     // tensor_dim1 = 128 (tile rows)
    g1[3] = (int)(32u << 16);                      // tile_dim0 = 32
    g1[4] = (int)128;                              // tile_dim1 = 128
    g1[5] = (int)F;                                // tensor_dim0_stride = F
    g1[6] = 0; g1[7] = 0;
    v4i_t gz4; gz4[0] = 0; gz4[1] = 0; gz4[2] = 0; gz4[3] = 0;
    v8i_t gz8;
#pragma unroll
    for (int q = 0; q < 8; ++q) gz8[q] = 0;
#else
    const int arow  = tid >> 1;
    const int ahalf = tid & 1;
    const uint4* asrc0 = (const uint4*)(hbuf + ((size_t)(e * C + m0 + arow)) * F + ahalf * 16);
    uint4 apre0, apre1;
#endif

#if !USE_TDM
    { const uint4* p = asrc0; apre0 = p[0]; apre1 = p[1]; }
#endif
    load_b(0);

    for (int kt = 0; kt < F / BK; ++kt) {
        __syncthreads();
#if USE_TDM
        if (wid == 0) {
            unsigned long long ga = (unsigned long long)(uintptr_t)(atile0 + kt * BK);
            v4u g0;
            g0[0] = 1u;                               // count=1
            g0[1] = lds_addr;
            g0[2] = (unsigned int)ga;
            g0[3] = (unsigned int)((ga >> 32) & 0x01FFFFFFu) | (2u << 30);  // type=2
            __builtin_amdgcn_tensor_load_to_lds(g0, g1, gz4, gz4, gz8, 0);
        }
        store_b();
        if (wid == 0) __builtin_amdgcn_s_wait_tensorcnt(0);
#else
        {
            uint4* ad = (uint4*)(&As[arow * LDT + ahalf * 16]);
            ad[0] = apre0; ad[1] = apre1;
        }
        store_b();
#endif
        __syncthreads();
        if (kt + 1 < F / BK) {
#if !USE_TDM
            const uint4* p = (const uint4*)((const unsigned short*)asrc0 + (kt + 1) * BK);
            apre0 = p[0]; apre1 = p[1];
#endif
            load_b(kt + 1);
        }

        v16bf afrag[2], bfrag[4];
        load_frags(As, Bs, wm, wn, grp, lr, afrag, bfrag);
#pragma unroll
        for (int i = 0; i < 2; ++i)
#pragma unroll
            for (int j = 0; j < 4; ++j)
                acc[i][j] = __builtin_amdgcn_wmma_f32_16x16x32_bf16(
                    false, afrag[i], false, bfrag[j], (short)0, acc[i][j], false, false);
    }

    // epilogue: bias, gate scale, scatter to out[token]
#pragma unroll
    for (int i = 0; i < 2; ++i)
#pragma unroll
        for (int j = 0; j < 4; ++j) {
            int col = n0 + wn * 64 + j * 16 + lr;
            float bias = b2[e * H + col];
#pragma unroll
            for (int rj = 0; rj < 8; ++rj) {
                int slot  = m0 + wm * 32 + i * 16 + grp * 8 + rj;
                int token = tok_of_slot[e * C + slot];
                if (token >= 0) {
                    float gate = gate_of_slot[e * C + slot];
                    out[(size_t)token * H + col] = gate * (acc[i][j][rj] + bias);
                }
            }
        }
}

// ---------------- kernel 5: l_aux + exp_counts ----------------
__global__ void moe_finalize_kernel(const float* __restrict__ stats,
                                    float* __restrict__ out) {
    const float* me  = stats;
    const int*   cnt = (const int*)(stats + 8);
    const size_t base = (size_t)S * H;
    if (threadIdx.x == 0) {
        float l = 0.0f;
        for (int e = 0; e < E; ++e)
            l += (me[e] / (float)S) * ((float)cnt[e] / (float)S);
        out[base] = l * (float)E;
    }
    if (threadIdx.x < E) out[base + 1 + threadIdx.x] = (float)cnt[threadIdx.x];
}

// ---------------- host launcher ----------------
extern "C" void kernel_launch(void* const* d_in, const int* in_sizes, int n_in,
                              void* d_out, int out_size, void* d_ws, size_t ws_size,
                              hipStream_t stream) {
    const float* x  = (const float*)d_in[0];
    const float* wg = (const float*)d_in[1];
    const float* w1 = (const float*)d_in[2];
    const float* b1 = (const float*)d_in[3];
    const float* w2 = (const float*)d_in[4];
    const float* b2 = (const float*)d_in[5];
    float* out = (float*)d_out;

    char* ws = (char*)d_ws;
    size_t off = 0;
    auto carve = [&](size_t bytes) { void* p = ws + off; off = (off + bytes + 255) & ~(size_t)255; return p; };
    int*            idx1      = (int*)carve((size_t)S * 4);
    float*          gate_max  = (float*)carve((size_t)S * 4);
    int*            tok       = (int*)carve((size_t)E * C * 4);
    float*          gslot     = (float*)carve((size_t)E * C * 4);
    float*          stats     = (float*)carve(64);
    unsigned short* hbuf      = (unsigned short*)carve((size_t)E * C * F * 2);
    (void)ws_size; (void)n_in; (void)in_sizes;

    {
        int blocks = (out_size + 1023) / 1024;
        moe_zero_kernel<<<blocks, 1024, 0, stream>>>(out, out_size, (unsigned int*)stats);
    }
    moe_gate_kernel<<<S / 8, 256, 0, stream>>>(x, wg, idx1, gate_max, stats);
    moe_scan_kernel<<<E, 1024, 0, stream>>>(idx1, gate_max, tok, gslot);
    {
        dim3 grid(F / BN, C / BM, E);
        moe_gemm1_kernel<<<grid, 256, 0, stream>>>(x, w1, b1, tok, hbuf);
    }
    {
        dim3 grid(H / BN, C / BM, E);
        moe_gemm2_kernel<<<grid, 256, 0, stream>>>(hbuf, w2, b2, tok, gslot, out);
    }
    moe_finalize_kernel<<<1, 32, 0, stream>>>(stats, out);
}
